// BiLSTMLayer_52544629899489
// MI455X (gfx1250) — compile-verified, matrix operand order checked by reading, not run on hardware
//
#include <hip/hip_runtime.h>
#include <math.h>

// ---------------------------------------------------------------------------
// MI455X (gfx1250) fused bi-LSTM.
//  * All GEMMs via v_wmma_f32_16x16x32_bf16 (wave32, 16x16 tiles, f32 accum).
//  * Each wave owns 16 batch rows and scans the full sequence for one
//    layer, both directions in turn; gates kept in WMMA C/D fragments so the LSTM cell
//    update is elementwise on matching VGPR components.
//  * Gate activations use gfx1250 v_tanh_f32 (TRANS, co-executes with WMMA).
//  * A fragments = 2x ds_load_b128 per 16x32 tile (layout is 16B/lane/half).
//  * All LDS regions are per-wave; same-wave DS ordering -> no barriers.
//  * Weights pre-packed into B-fragment order (bf16) in scratch each call.
//  * ws layout: [0,134MB) bf16 activation ping; then packed weights (~0.6MB).
//    d_out doubles as the bf16 activation pong before the final LN pass.
// ---------------------------------------------------------------------------

typedef __attribute__((ext_vector_type(16))) __bf16 v16bf;
typedef __attribute__((ext_vector_type(8)))  float  v8f;

#define H_    64
#define T_    64
#define BATCH 8192
#define D2_   128
#define G4_   256

#define LDW_IN 136   // sh_in row stride (elems): 272B = 17*16, bank stride 68
#define LDW_H  88    // sh_h  row stride (elems): 176B = 11*16, bank stride 44

#define FRAG_ELEMS 512                 // 32 lanes * 16 bf16 per 32x16 B fragment
#define WIH_LD (16 * 4 * FRAG_ELEMS)   // per (layer,dir): 16 ntiles * 4 ktiles
#define WHH_LD (16 * 2 * FRAG_ELEMS)   // per (layer,dir): 16 ntiles * 2 ktiles

// K index held by (VGPR r, lane-half hi) for a 16-bit 16x32 A fragment
// (ISA 7.12.2): lanes0-15 V0..3 -> K0..7, V4..7 -> K16..23; lanes16-31 +8.
// B fragments are packed with the mirrored mapping by the pack kernel.
__device__ __host__ __forceinline__ int kmap(int r, int hi) {
  return ((r & 3) << 1) + (hi << 3) + ((r >> 2) << 4);
}

// gfx1250 has V_TANH_F32 (transcendental, co-executes with WMMA).
__device__ __forceinline__ float dev_tanh(float x) {
#if defined(__AMDGCN__) && __has_builtin(__builtin_amdgcn_tanhf)
  return __builtin_amdgcn_tanhf(x);
#elif defined(__AMDGCN__) && __has_builtin(__builtin_amdgcn_tanh_f32)
  return __builtin_amdgcn_tanh_f32(x);
#elif defined(__AMDGCN__)
  float e2 = __builtin_amdgcn_exp2f(x * 2.8853900817779268f);
  return 1.f - 2.f * __builtin_amdgcn_rcpf(e2 + 1.f);
#else
  return tanhf(x);
#endif
}
__device__ __forceinline__ float dev_sig(float x) {
  return fmaf(0.5f, dev_tanh(0.5f * x), 0.5f);
}

__device__ __forceinline__ v16bf frag2(uint4 a, uint4 b) {
  union { uint4 u[2]; v16bf v; } cv;
  cv.u[0] = a;
  cv.u[1] = b;
  return cv.v;
}

// 16x32 bf16 A fragment from a row-major LDS tile with 16B-aligned row stride:
// lane (n, hi) takes 16B at [n][kbase + hi*8] (V0..3) and +16 elems (V4..7).
__device__ __forceinline__ v16bf load_afrag(const __bf16* rowp, int kbase, int hi) {
  uint4 lo = *reinterpret_cast<const uint4*>(rowp + kbase + hi * 8);
  uint4 hi4 = *reinterpret_cast<const uint4*>(rowp + kbase + 16 + hi * 8);
  return frag2(lo, hi4);
}

// B fragment: SGPR base + 32-bit element offset (keeps addressing in GVS form).
__device__ __forceinline__ v16bf load_bfrag(const __bf16* __restrict__ base,
                                            unsigned off_elems) {
  const uint4* q = reinterpret_cast<const uint4*>(base + off_elems);
  return frag2(q[0], q[1]);
}

// ---------------------------------------------------------------------------
// Pack w_ih / w_hh into WMMA B-fragment order (bf16) + fused bias.
// ---------------------------------------------------------------------------
__global__ void pack_weights_kernel(const float* __restrict__ w_ih,
                                    const float* __restrict__ w_hh,
                                    const float* __restrict__ b_ih,
                                    const float* __restrict__ b_hh,
                                    __bf16* __restrict__ wihP,
                                    __bf16* __restrict__ whhP,
                                    float* __restrict__ biasP) {
  const int NIH = 6 * WIH_LD;
  const int NHH = 6 * WHH_LD;
  int idx = blockIdx.x * blockDim.x + threadIdx.x;
  if (idx < NIH) {
    int ld   = idx / WIH_LD;
    int rem  = idx % WIH_LD;
    int nt   = rem / (4 * FRAG_ELEMS);
    int rem2 = rem % (4 * FRAG_ELEMS);
    int kt   = rem2 / FRAG_ELEMS;
    int rem3 = rem2 % FRAG_ELEMS;
    int lane = rem3 >> 4, e = rem3 & 15;
    int r = e >> 1, p = e & 1, hi = lane >> 4, n = lane & 15;
    int k = kt * 32 + kmap(r, hi) + p;              // input-dim index (0..127)
    float v = w_ih[(size_t)ld * G4_ * D2_ + (size_t)(nt * 16 + n) * D2_ + k];
    wihP[idx] = (__bf16)v;
  } else if (idx < NIH + NHH) {
    int id2  = idx - NIH;
    int ld   = id2 / WHH_LD;
    int rem  = id2 % WHH_LD;
    int nt   = rem / (2 * FRAG_ELEMS);
    int rem2 = rem % (2 * FRAG_ELEMS);
    int kt   = rem2 / FRAG_ELEMS;
    int rem3 = rem2 % FRAG_ELEMS;
    int lane = rem3 >> 4, e = rem3 & 15;
    int r = e >> 1, p = e & 1, hi = lane >> 4, n = lane & 15;
    int k = kt * 32 + kmap(r, hi) + p;              // hidden index (0..63)
    float v = w_hh[(size_t)ld * G4_ * H_ + (size_t)(nt * 16 + n) * H_ + k];
    whhP[id2] = (__bf16)v;
  } else if (idx < NIH + NHH + 6 * G4_) {
    int id3 = idx - NIH - NHH;
    biasP[id3] = b_ih[id3] + b_hh[id3];
  }
}

// ---------------------------------------------------------------------------
// One bi-LSTM layer. Grid: 64 blocks x 256 threads (8 waves). Each wave:
// 16 batch rows, full T scan, both directions sequentially.
// ---------------------------------------------------------------------------
__global__ __launch_bounds__(256) void lstm_layer_kernel(
    const float*  __restrict__ x0,     // layer0 input (B,T,8) or null
    const __bf16* __restrict__ inAct,  // (B,T,128) bf16 or null
    __bf16*       __restrict__ outAct, // (B,T,128) bf16
    const __bf16* __restrict__ wihL,   // this layer, 2 dirs, fragment order
    const __bf16* __restrict__ whhL,
    const float*  __restrict__ biasL,  // this layer, 2 dirs, b_ih+b_hh
    int layer0) {
  __shared__ __align__(16) __bf16 sh_in[8][16][LDW_IN]; // per-wave input tile
  __shared__ __align__(16) __bf16 sh_h[8][16][LDW_H];   // per-wave h tile

  const int wave  = threadIdx.x >> 5;
  const int lane  = threadIdx.x & 31;
  const int hi    = lane >> 4;
  const int n     = lane & 15;          // column (N) / A-row (M) index
  const int bbase = blockIdx.x * 128 + wave * 16;
  const int nkt   = layer0 ? 1 : 4;     // layer0: only K 0..31 nonzero (Din=8)
  const unsigned laneOff = (unsigned)lane * 16;

  for (int dir = 0; dir < 2; ++dir) {
    const __bf16* wih  = wihL + dir * WIH_LD;
    const __bf16* whh  = whhL + dir * WHH_LD;
    const float*  bias = biasL + dir * G4_;

    // preload fused biases for this lane's column (hoisted out of the scan)
    float bv[4][4];
#pragma unroll
    for (int g = 0; g < 4; ++g)
#pragma unroll
      for (int j = 0; j < 4; ++j) bv[g][j] = bias[g * 64 + j * 16 + n];

    // zero h stage + c state
    for (int i = lane; i < 16 * (LDW_H / 4); i += 32) {
      int rr = i / (LDW_H / 4), c4 = (i % (LDW_H / 4)) * 4;
      *reinterpret_cast<uint2*>(&sh_h[wave][rr][c4]) = make_uint2(0u, 0u);
    }
    float cst[4][8];
#pragma unroll
    for (int j = 0; j < 4; ++j)
#pragma unroll
      for (int r = 0; r < 8; ++r) cst[j][r] = 0.f;

    for (int t = 0; t < T_; ++t) {
      const int tt = dir ? (T_ - 1 - t) : t;

      // ---- stage input tile into LDS (bf16) ----
      if (layer0) {
        for (int i = lane; i < 16 * 32; i += 32) {
          int rr = i >> 5, cc = i & 31;
          float v = (cc < 8) ? x0[((size_t)(bbase + rr) * T_ + tt) * 8 + cc] : 0.f;
          sh_in[wave][rr][cc] = (__bf16)v;
        }
      } else {
        for (int i = lane; i < 16 * 32; i += 32) {   // 8-byte chunks
          int rr = i >> 5, c4 = (i & 31) * 4;
          uint2 u = *reinterpret_cast<const uint2*>(
              inAct + ((size_t)(bbase + rr) * T_ + tt) * D2_ + c4);
          *reinterpret_cast<uint2*>(&sh_in[wave][rr][c4]) = u;
        }
        if (t + 1 < T_) {   // prefetch next timestep's rows (global_prefetch_b8)
          int tn = dir ? (T_ - 2 - t) : (t + 1);
          __builtin_prefetch(inAct + ((size_t)(bbase + n) * T_ + tn) * D2_, 0, 0);
        }
      }

      // ---- gather A fragments: 2x b128 per 16x32 tile ----
      const __bf16* inrow = &sh_in[wave][n][0];
      const __bf16* hrow  = &sh_h[wave][n][0];
      v16bf a_in[4];
#pragma unroll
      for (int kt = 0; kt < 4; ++kt)
        if (kt < nkt) a_in[kt] = load_afrag(inrow, kt * 32, hi);
      v16bf a_h[2];
#pragma unroll
      for (int kt = 0; kt < 2; ++kt) a_h[kt] = load_afrag(hrow, kt * 32, hi);

      // ---- gates, 4 hidden-column groups of 16 ----
#pragma unroll
      for (int j = 0; j < 4; ++j) {
        v8f acc[4];   // i, f, g, o tiles (gate g -> ntile g*4+j)
#pragma unroll
        for (int g = 0; g < 4; ++g)
#pragma unroll
          for (int r = 0; r < 8; ++r) acc[g][r] = bv[g][j];
#pragma unroll
        for (int g = 0; g < 4; ++g) {
          unsigned nt = (unsigned)(g * 4 + j);
          for (int kt = 0; kt < nkt; ++kt) {
            v16bf b = load_bfrag(wih, (nt * 4 + kt) * FRAG_ELEMS + laneOff);
            acc[g] = __builtin_amdgcn_wmma_f32_16x16x32_bf16(
                false, a_in[kt], false, b, (short)0, acc[g], false, false);
          }
#pragma unroll
          for (int kt = 0; kt < 2; ++kt) {
            v16bf b = load_bfrag(whh, (nt * 2 + kt) * FRAG_ELEMS + laneOff);
            acc[g] = __builtin_amdgcn_wmma_f32_16x16x32_bf16(
                false, a_h[kt], false, b, (short)0, acc[g], false, false);
          }
        }
        // ---- LSTM cell update (elementwise on matching fragment slots) ----
#pragma unroll
        for (int r = 0; r < 8; ++r) {
          float ig = dev_sig(acc[0][r]);
          float fg = dev_sig(acc[1][r]);
          float gg = dev_tanh(acc[2][r]);
          float og = dev_sig(acc[3][r]);
          float cv = fmaf(fg, cst[j][r], ig * gg);
          cst[j][r] = cv;
          float hv = og * dev_tanh(cv);
          sh_h[wave][r + hi * 8][j * 16 + n] = (__bf16)hv;
        }
      }

      // ---- vectorized store of the finished h tile (16x64) to global ----
      {
        int r = lane & 15, half = lane >> 4;
        const uint4* src = reinterpret_cast<const uint4*>(&sh_h[wave][r][half * 32]);
        uint4* dst = reinterpret_cast<uint4*>(
            outAct + ((size_t)(bbase + r) * T_ + tt) * D2_ + dir * H_ + half * 32);
#pragma unroll
        for (int q2 = 0; q2 < 4; ++q2) dst[q2] = src[q2];
      }
    }
  }
}

// ---------------------------------------------------------------------------
// relu(lstm) + x @ w_res^T + b_res, then LayerNorm over 128 features.
// One wave per (b,t) row; 4 features per lane; __shfl_xor wave32 reductions.
// ---------------------------------------------------------------------------
__global__ __launch_bounds__(256) void ln_res_kernel(
    const __bf16* __restrict__ act, const float* __restrict__ x,
    const float* __restrict__ w_res, const float* __restrict__ b_res,
    const float* __restrict__ gamma, const float* __restrict__ beta,
    float* __restrict__ out) {
  const int row  = blockIdx.x * 8 + (threadIdx.x >> 5);
  const int lane = threadIdx.x & 31;
  if (row >= BATCH * T_) return;
  const float* xr = x + (size_t)row * 8;
  float xd[8];
#pragma unroll
  for (int d = 0; d < 8; ++d) xd[d] = xr[d];
  union { uint2 u; __bf16 h[4]; } av;
  av.u = *reinterpret_cast<const uint2*>(act + (size_t)row * D2_ + lane * 4);
  float v[4];
#pragma unroll
  for (int q = 0; q < 4; ++q) {
    int f = lane * 4 + q;
    float res = b_res[f];
#pragma unroll
    for (int d = 0; d < 8; ++d) res = fmaf(xd[d], w_res[f * 8 + d], res);
    float a = fmaxf((float)av.h[q], 0.f);
    v[q] = a + res;
  }
  float s = v[0] + v[1] + v[2] + v[3];
#pragma unroll
  for (int off = 16; off > 0; off >>= 1) s += __shfl_xor(s, off, 32);
  float mu = s * (1.f / 128.f);
  float sq = 0.f;
#pragma unroll
  for (int q = 0; q < 4; ++q) { float d = v[q] - mu; sq += d * d; }
#pragma unroll
  for (int off = 16; off > 0; off >>= 1) sq += __shfl_xor(sq, off, 32);
  float rs = rsqrtf(sq * (1.f / 128.f) + 1e-5f);
#pragma unroll
  for (int q = 0; q < 4; ++q) {
    int f = lane * 4 + q;
    out[(size_t)row * D2_ + f] = (v[q] - mu) * rs * gamma[f] + beta[f];
  }
}

extern "C" void kernel_launch(void* const* d_in, const int* in_sizes, int n_in,
                              void* d_out, int out_size, void* d_ws, size_t ws_size,
                              hipStream_t stream) {
  const float* x     = (const float*)d_in[0];
  const float* w_ih  = (const float*)d_in[1];
  const float* w_hh  = (const float*)d_in[2];
  const float* b_ih  = (const float*)d_in[3];
  const float* b_hh  = (const float*)d_in[4];
  const float* w_res = (const float*)d_in[5];
  const float* b_res = (const float*)d_in[6];
  const float* gamma = (const float*)d_in[7];
  const float* beta  = (const float*)d_in[8];
  float* out = (float*)d_out;

  // ws: [0, 134MB) bf16 activation ping-buffer; then packed weights (~0.6MB).
  char* ws = (char*)d_ws;
  const size_t actBytes = (size_t)BATCH * T_ * D2_ * sizeof(__bf16);
  __bf16* actW  = (__bf16*)ws;
  __bf16* wihP  = (__bf16*)(ws + actBytes);
  __bf16* whhP  = wihP + 6 * WIH_LD;
  float*  biasP = (float*)(whhP + 6 * WHH_LD);
  __bf16* actO  = (__bf16*)d_out;   // pong: reuse d_out bytes before final LN

  {
    int total = 6 * WIH_LD + 6 * WHH_LD + 6 * G4_;
    pack_weights_kernel<<<(total + 255) / 256, 256, 0, stream>>>(
        w_ih, w_hh, b_ih, b_hh, wihP, whhP, biasP);
  }
  dim3 lgrid(BATCH / 128), lblk(256);
  lstm_layer_kernel<<<lgrid, lblk, 0, stream>>>(
      x, nullptr, actW, wihP + 0 * 2 * WIH_LD, whhP + 0 * 2 * WHH_LD,
      biasP + 0 * 2 * G4_, 1);
  lstm_layer_kernel<<<lgrid, lblk, 0, stream>>>(
      nullptr, actW, actO, wihP + 1 * 2 * WIH_LD, whhP + 1 * 2 * WHH_LD,
      biasP + 1 * 2 * G4_, 0);
  lstm_layer_kernel<<<lgrid, lblk, 0, stream>>>(
      nullptr, actO, actW, wihP + 2 * 2 * WIH_LD, whhP + 2 * 2 * WHH_LD,
      biasP + 2 * 2 * G4_, 0);
  ln_res_kernel<<<(BATCH * T_) / 8, 256, 0, stream>>>(
      actW, x, w_res, b_res, gamma, beta, out);
}
